// FeatureNet_69587060129986
// MI455X (gfx1250) — matrix operand order repaired and checked by reading, not compiled, not run on hardware
//
#include <hip/hip_runtime.h>
#include <hip/hip_bf16.h>

// ---------------------------------------------------------------------------
// FeatureNet (VoxelNet VFE x2 + dense) for MI455X / gfx1250.
//
// Pipeline (batch-stat BN forces global syncs between matmuls):
//   1. k_layer1      : x[7] -> pw1[16] (VALU fp32, tiny K) + mask
//   2. k_stats<16>   : per-channel sum/sumsq of pw1 (deterministic tree)
//   3. k_finalize    : BN1 scale/shift
//   4. k_agg<16>     : agg1[k,c] = max_t relu(bn1(pw1))
//   5. k_gemm2       : h1[32] = concat(relu(bn1(pw1)),agg1)*mask ; WMMA bf16
//                      16x16x32 (K=32 == inner dim, one WMMA per tile) -> pw2
//   6. k_stats<64> / k_finalize / k_agg<64>
//   7. k_gemm3_stats : h2[128] @ Wd via 4x WMMA bf16 chained; accumulates
//                      sum/sumsq of hd WITHOUT storing hd (saves ~1 GB HBM)
//   8. k_finalize    : BNd scale/shift
//   9. k_gemm3_out   : recompute hd per-voxel (35 rows -> 3 16-row tiles),
//                      BN + leaky relu + max-over-T inside the wave
//                      (shfl_xor(16)), direct store to out[128][K]
//  10. k_coord      : bit-copy int coordinates into output tail
// No atomics anywhere -> bit-deterministic across replays.
// ---------------------------------------------------------------------------

#define KVOX 30000
#define TPT  35
#define NCIN 7
#define NROWS (KVOX * TPT)        // 1,050,000 (divisible by 16)
#define NTILES (NROWS / 16)       // 65,625
#define BN_EPS 1e-5f

typedef __attribute__((ext_vector_type(16))) __bf16 bf16x16;
typedef __attribute__((ext_vector_type(8)))  float  f32x8;

static_assert(NROWS % 16 == 0, "row count must tile by 16");

__device__ __forceinline__ float4 ldg4(const float* p) {
    return *reinterpret_cast<const float4*>(p);
}

__device__ __forceinline__ float4 bnrelu4(float4 v, float4 sc, float4 sh, float mk) {
    return make_float4(fmaxf(fmaf(v.x, sc.x, sh.x), 0.f) * mk,
                       fmaxf(fmaf(v.y, sc.y, sh.y), 0.f) * mk,
                       fmaxf(fmaf(v.z, sc.z, sh.z), 0.f) * mk,
                       fmaxf(fmaf(v.w, sc.w, sh.w), 0.f) * mk);
}

__device__ __forceinline__ float4 mul4(float4 v, float m) {
    return make_float4(v.x * m, v.y * m, v.z * m, v.w * m);
}

__device__ __forceinline__ void pack8(bf16x16& v, int base, float4 a, float4 b) {
    v[base + 0] = (__bf16)a.x; v[base + 1] = (__bf16)a.y;
    v[base + 2] = (__bf16)a.z; v[base + 3] = (__bf16)a.w;
    v[base + 4] = (__bf16)b.x; v[base + 5] = (__bf16)b.y;
    v[base + 6] = (__bf16)b.z; v[base + 7] = (__bf16)b.w;
}

// ---------------------------------------------------------------------------
// 1) layer 1: x[r,0..6] @ W1(7x16) + b1 -> pw1 ; mask[r] = (sum(x) != 0)
// ---------------------------------------------------------------------------
__global__ void k_layer1(const float* __restrict__ x, const float* __restrict__ W1,
                         const float* __restrict__ b1, float* __restrict__ pw1,
                         float* __restrict__ mask) {
    int r = blockIdx.x * blockDim.x + threadIdx.x;
    if (r >= NROWS) return;
    float xv[NCIN];
    float s = 0.f;
#pragma unroll
    for (int i = 0; i < NCIN; i++) { xv[i] = x[(size_t)r * NCIN + i]; s += xv[i]; }
    mask[r] = (s != 0.f) ? 1.f : 0.f;
#pragma unroll
    for (int c = 0; c < 16; c++) {
        float acc = b1[c];
#pragma unroll
        for (int i = 0; i < NCIN; i++) acc = fmaf(xv[i], W1[i * 16 + c], acc);
        pw1[(size_t)r * 16 + c] = acc;
    }
}

// ---------------------------------------------------------------------------
// per-channel sum / sumsq -> partials[block][2C] (fixed-order, deterministic)
// ---------------------------------------------------------------------------
template <int C>
__global__ void k_stats(const float* __restrict__ d, float* __restrict__ part) {
    const int G = 256 / C;
    int t = threadIdx.x;
    int c = t % C, g = t / C;
    float s = 0.f, q = 0.f;
    for (int r = blockIdx.x * G + g; r < NROWS; r += gridDim.x * G) {
        float v = d[(size_t)r * C + c];
        s += v;
        q = fmaf(v, v, q);
    }
    __shared__ float sh[512];
    sh[t] = s; sh[256 + t] = q;
    __syncthreads();
    if (t < C) {
        float S = 0.f, Q = 0.f;
        for (int gg = 0; gg < G; gg++) { S += sh[gg * C + t]; Q += sh[256 + gg * C + t]; }
        part[(size_t)blockIdx.x * (2 * C) + t]     = S;
        part[(size_t)blockIdx.x * (2 * C) + C + t] = Q;
    }
}

// partials -> bn params:  bnp[c]=gamma*rsqrt(var+eps), bnp[C+c]=beta-mean*scale
__global__ void k_finalize(const float* __restrict__ part, int nP, int C,
                           const float* __restrict__ gamma, const float* __restrict__ beta,
                           float* __restrict__ bnp) {
    int c = threadIdx.x;
    if (c >= C) return;
    float s = 0.f, q = 0.f;
    for (int p = 0; p < nP; p++) { s += part[(size_t)p * 2 * C + c]; q += part[(size_t)p * 2 * C + C + c]; }
    const float invN = 1.0f / (float)NROWS;
    float mean = s * invN;
    float var  = q * invN - mean * mean;
    float sc   = gamma[c] * rsqrtf(var + BN_EPS);
    bnp[c]     = sc;
    bnp[C + c] = beta[c] - mean * sc;
}

// agg[k,c] = max_t relu(bn(pw[k,t,c]))
template <int C>
__global__ void k_agg(const float* __restrict__ pw, const float* __restrict__ bnp,
                      float* __restrict__ agg) {
    int tid = blockIdx.x * blockDim.x + threadIdx.x;
    int k = tid / C, c = tid % C;
    if (k >= KVOX) return;
    float sc = bnp[c], sh = bnp[C + c];
    float m = 0.f;  // relu output >= 0
    for (int t = 0; t < TPT; t++) {
        float v = fmaxf(fmaf(pw[((size_t)k * TPT + t) * C + c], sc, sh), 0.f);
        m = fmaxf(m, v);
    }
    agg[(size_t)k * C + c] = m;
}

// ---------------------------------------------------------------------------
// 5) GEMM2: h1[M,32] @ W2[32,64] + b2 -> pw2[M,64]  (1 WMMA per 16x16 tile)
// ---------------------------------------------------------------------------
__global__ void __launch_bounds__(256) k_gemm2(
    const float* __restrict__ pw1, const float* __restrict__ agg1,
    const float* __restrict__ mask, const float* __restrict__ bn1p,
    const float* __restrict__ W2, const float* __restrict__ b2,
    float* __restrict__ pw2) {
    __shared__ __bf16 sW2[64 * 40];  // [u][i], padded stride 40 halves (80B)
    for (int idx = threadIdx.x; idx < 32 * 64; idx += blockDim.x) {
        int i = idx >> 6, u = idx & 63;
        sW2[u * 40 + i] = (__bf16)W2[idx];
    }
    __syncthreads();

    const int lane = threadIdx.x & 31;
    const int lo = lane & 15;
    const int hi = lane >> 4;   // 0: K 0-7/16-23, 1: K 8-15/24-31
    const int h8 = hi * 8;

    // B fragments: constant for the whole kernel, hoisted
    bf16x16 bfr[4];
    float b2v[4];
#pragma unroll
    for (int nt = 0; nt < 4; nt++) {
        int u = nt * 16 + lo;
        __builtin_memcpy(&bfr[nt], &sW2[u * 40 + h8], 16);
        __builtin_memcpy((char*)&bfr[nt] + 16, &sW2[u * 40 + 16 + h8], 16);
        b2v[nt] = b2[u];
    }

    const int wave  = (blockIdx.x * blockDim.x + threadIdx.x) >> 5;
    const int nwave = (gridDim.x * blockDim.x) >> 5;

    float4 sc0 = ldg4(bn1p + h8),      sc1 = ldg4(bn1p + h8 + 4);
    float4 sh0 = ldg4(bn1p + 16 + h8), sh1 = ldg4(bn1p + 16 + h8 + 4);

    for (int tile = wave; tile < NTILES; tile += nwave) {
        int r = tile * 16 + lo;     // A-matrix row for this lane
        int k = r / TPT;
        float mk = mask[r];
        float4 p0 = ldg4(pw1 + (size_t)r * 16 + h8);
        float4 p1 = ldg4(pw1 + (size_t)r * 16 + h8 + 4);
        float4 a0 = ldg4(agg1 + (size_t)k * 16 + h8);
        float4 a1 = ldg4(agg1 + (size_t)k * 16 + h8 + 4);
        bf16x16 afr;
        pack8(afr, 0, bnrelu4(p0, sc0, sh0, mk), bnrelu4(p1, sc1, sh1, mk));
        pack8(afr, 8, mul4(a0, mk), mul4(a1, mk));
#pragma unroll
        for (int nt = 0; nt < 4; nt++) {
            f32x8 acc = {};
            acc = __builtin_amdgcn_wmma_f32_16x16x32_bf16(
                false, afr, false, bfr[nt], (short)0, acc, false, false);
            int ubase = nt * 16 + lo;
#pragma unroll
            for (int j = 0; j < 8; j++) {
                int row = tile * 16 + j + h8;   // C-matrix row mapping
                pw2[(size_t)row * 64 + ubase] = acc[j] + b2v[nt];
            }
        }
    }
}

// ---------------------------------------------------------------------------
// h2 A-fragment builder: h2[i] = i<64 ? relu(bn2(pw2))*mk : agg2*mk
// frags ks=0..1 from pw2, ks=2..3 from agg2
// ---------------------------------------------------------------------------
__device__ __forceinline__ void build_h2_frags(
    bf16x16 (&af)[4], const float* __restrict__ pw2, const float* __restrict__ agg2,
    const float* __restrict__ bn2p, int r, int k, float mk, int h8) {
#pragma unroll
    for (int ks = 0; ks < 2; ks++) {
        int cA = ks * 32 + h8;
        int cB = ks * 32 + 16 + h8;
        float4 pa0 = ldg4(pw2 + (size_t)r * 64 + cA);
        float4 pa1 = ldg4(pw2 + (size_t)r * 64 + cA + 4);
        float4 pb0 = ldg4(pw2 + (size_t)r * 64 + cB);
        float4 pb1 = ldg4(pw2 + (size_t)r * 64 + cB + 4);
        float4 scA0 = ldg4(bn2p + cA),      scA1 = ldg4(bn2p + cA + 4);
        float4 shA0 = ldg4(bn2p + 64 + cA), shA1 = ldg4(bn2p + 64 + cA + 4);
        float4 scB0 = ldg4(bn2p + cB),      scB1 = ldg4(bn2p + cB + 4);
        float4 shB0 = ldg4(bn2p + 64 + cB), shB1 = ldg4(bn2p + 64 + cB + 4);
        pack8(af[ks], 0, bnrelu4(pa0, scA0, shA0, mk), bnrelu4(pa1, scA1, shA1, mk));
        pack8(af[ks], 8, bnrelu4(pb0, scB0, shB0, mk), bnrelu4(pb1, scB1, shB1, mk));
    }
#pragma unroll
    for (int ks = 0; ks < 2; ks++) {
        int cA = ks * 32 + h8;
        int cB = ks * 32 + 16 + h8;
        float4 a0 = ldg4(agg2 + (size_t)k * 64 + cA);
        float4 a1 = ldg4(agg2 + (size_t)k * 64 + cA + 4);
        float4 b0 = ldg4(agg2 + (size_t)k * 64 + cB);
        float4 b1 = ldg4(agg2 + (size_t)k * 64 + cB + 4);
        pack8(af[2 + ks], 0, mul4(a0, mk), mul4(a1, mk));
        pack8(af[2 + ks], 8, mul4(b0, mk), mul4(b1, mk));
    }
}

#define WD_STRIDE 136  // halves; 272B row stride, 16B aligned, conflict-friendly

__device__ __forceinline__ void stage_wd(__bf16* sWd, const float* __restrict__ Wd) {
    for (int idx = threadIdx.x; idx < 128 * 128; idx += blockDim.x) {
        int c = idx >> 7, d = idx & 127;          // Wd[c][d]
        sWd[d * WD_STRIDE + c] = (__bf16)Wd[idx]; // store transposed [d][c]
    }
}

// ---------------------------------------------------------------------------
// 7) GEMM3 pass 1: per-channel sum/sumsq of hd = h2 @ Wd + bd (hd not stored)
// ---------------------------------------------------------------------------
__global__ void __launch_bounds__(256) k_gemm3_stats(
    const float* __restrict__ pw2, const float* __restrict__ agg2,
    const float* __restrict__ mask, const float* __restrict__ bn2p,
    const float* __restrict__ Wd, const float* __restrict__ bd,
    float* __restrict__ part) {
    __shared__ __bf16 sWd[128 * WD_STRIDE];
    __shared__ float  sRed[8 * 256];
    stage_wd(sWd, Wd);
    __syncthreads();

    const int lane = threadIdx.x & 31;
    const int lo = lane & 15, hi = lane >> 4, h8 = hi * 8;
    const int wid = threadIdx.x >> 5;
    const int wave  = (blockIdx.x * blockDim.x + threadIdx.x) >> 5;
    const int nwave = (gridDim.x * blockDim.x) >> 5;

    float sAcc[8] = {0, 0, 0, 0, 0, 0, 0, 0};
    float qAcc[8] = {0, 0, 0, 0, 0, 0, 0, 0};

    for (int tile = wave; tile < NTILES; tile += nwave) {
        int r = tile * 16 + lo;
        int k = r / TPT;
        float mk = mask[r];
        bf16x16 af[4];
        build_h2_frags(af, pw2, agg2, bn2p, r, k, mk, h8);
#pragma unroll
        for (int nt = 0; nt < 8; nt++) {
            int d = nt * 16 + lo;
            f32x8 acc = {};
#pragma unroll
            for (int ks = 0; ks < 4; ks++) {
                bf16x16 bf;
                __builtin_memcpy(&bf, &sWd[d * WD_STRIDE + ks * 32 + h8], 16);
                __builtin_memcpy((char*)&bf + 16, &sWd[d * WD_STRIDE + ks * 32 + 16 + h8], 16);
                acc = __builtin_amdgcn_wmma_f32_16x16x32_bf16(
                    false, af[ks], false, bf, (short)0, acc, false, false);
            }
            float bdv = bd[d];
#pragma unroll
            for (int j = 0; j < 8; j++) {
                float v = acc[j] + bdv;
                sAcc[nt] += v;
                qAcc[nt] = fmaf(v, v, qAcc[nt]);
            }
        }
    }
    // lane n <-> n+16 carry the same channel (different rows): combine, then
    // fixed-order cross-wave reduction in LDS -> per-block partials.
#pragma unroll
    for (int nt = 0; nt < 8; nt++) {
        float s = sAcc[nt] + __shfl_xor(sAcc[nt], 16, 32);
        float q = qAcc[nt] + __shfl_xor(qAcc[nt], 16, 32);
        if (lane < 16) {
            int ch = nt * 16 + lane;
            sRed[wid * 256 + ch]       = s;
            sRed[wid * 256 + 128 + ch] = q;
        }
    }
    __syncthreads();
    int t = threadIdx.x;
    float v = 0.f;
    for (int w = 0; w < 8; w++) v += sRed[w * 256 + t];
    part[(size_t)blockIdx.x * 256 + t] = v;   // [sum 0..127][sumsq 0..127]
}

// ---------------------------------------------------------------------------
// 9) GEMM3 pass 2: per-voxel recompute hd, BN + leaky relu, max over T,
//    direct store out[d*K + k]. T=35 -> 3 tiles of 16 rows (tail masked).
// ---------------------------------------------------------------------------
__global__ void __launch_bounds__(256) k_gemm3_out(
    const float* __restrict__ pw2, const float* __restrict__ agg2,
    const float* __restrict__ mask, const float* __restrict__ bn2p,
    const float* __restrict__ Wd, const float* __restrict__ bd,
    const float* __restrict__ bndp, float* __restrict__ out) {
    __shared__ __bf16 sWd[128 * WD_STRIDE];
    stage_wd(sWd, Wd);
    __syncthreads();

    const int lane = threadIdx.x & 31;
    const int lo = lane & 15, hi = lane >> 4, h8 = hi * 8;
    const int wave  = (blockIdx.x * blockDim.x + threadIdx.x) >> 5;
    const int nwave = (gridDim.x * blockDim.x) >> 5;

    for (int k = wave; k < KVOX; k += nwave) {
        bf16x16 af[3][4];
#pragma unroll
        for (int tau = 0; tau < 3; tau++) {
            int t = tau * 16 + lo;                 // A-matrix row = lane&15
            int valid = (t < TPT);
            int r = k * TPT + (valid ? t : 0);     // clamp; mk=0 zeroes frag
            float mk = valid ? mask[r] : 0.f;
            build_h2_frags(af[tau], pw2, agg2, bn2p, r, k, mk, h8);
        }
#pragma unroll
        for (int nt = 0; nt < 8; nt++) {
            int d = nt * 16 + lo;
            bf16x16 bf[4];
#pragma unroll
            for (int ks = 0; ks < 4; ks++) {
                __builtin_memcpy(&bf[ks], &sWd[d * WD_STRIDE + ks * 32 + h8], 16);
                __builtin_memcpy((char*)&bf[ks] + 16, &sWd[d * WD_STRIDE + ks * 32 + 16 + h8], 16);
            }
            float bdv = bd[d], sc = bndp[d], sh = bndp[128 + d];
            float m = -3.4e38f;
#pragma unroll
            for (int tau = 0; tau < 3; tau++) {
                f32x8 acc = {};
#pragma unroll
                for (int ks = 0; ks < 4; ks++)
                    acc = __builtin_amdgcn_wmma_f32_16x16x32_bf16(
                        false, af[tau][ks], false, bf[ks], (short)0, acc, false, false);
#pragma unroll
                for (int j = 0; j < 8; j++) {
                    int t = tau * 16 + j + h8;     // C-matrix row mapping
                    if (t < TPT) {
                        float v = fmaf(acc[j] + bdv, sc, sh);
                        v = (v >= 0.f) ? v : 0.1f * v;  // leaky relu(0.1)
                        m = fmaxf(m, v);
                    }
                }
            }
            m = fmaxf(m, __shfl_xor(m, 16, 32));   // merge lane n / n+16 rows
            if (lane < 16) out[(size_t)d * KVOX + k] = m;
        }
    }
}

__global__ void k_coord(const int* __restrict__ coord, int* __restrict__ dst) {
    int i = blockIdx.x * blockDim.x + threadIdx.x;
    if (i < KVOX * 4) dst[i] = coord[i];
}

// ---------------------------------------------------------------------------
// host side
// ---------------------------------------------------------------------------
extern "C" void kernel_launch(void* const* d_in, const int* in_sizes, int n_in,
                              void* d_out, int out_size, void* d_ws, size_t ws_size,
                              hipStream_t stream) {
    (void)in_sizes; (void)n_in; (void)out_size; (void)ws_size;
    const float* x   = (const float*)d_in[0];
    const int*   crd = (const int*)d_in[1];
    const float* W1  = (const float*)d_in[3];
    const float* b1  = (const float*)d_in[4];
    const float* g1  = (const float*)d_in[5];
    const float* be1 = (const float*)d_in[6];
    const float* W2  = (const float*)d_in[7];
    const float* b2  = (const float*)d_in[8];
    const float* g2  = (const float*)d_in[9];
    const float* be2 = (const float*)d_in[10];
    const float* Wd  = (const float*)d_in[11];
    const float* bd  = (const float*)d_in[12];
    const float* gd  = (const float*)d_in[13];
    const float* bed = (const float*)d_in[14];

    float* ws   = (float*)d_ws;
    float* pw1  = ws;                               // NROWS*16
    float* pw2  = pw1 + (size_t)NROWS * 16;         // NROWS*64
    float* msk  = pw2 + (size_t)NROWS * 64;         // NROWS
    float* agg1 = msk + NROWS;                      // KVOX*16
    float* agg2 = agg1 + (size_t)KVOX * 16;         // KVOX*64
    float* part = agg2 + (size_t)KVOX * 64;         // 1024*256 max
    float* bn1p = part + 1024 * 256;                // 32
    float* bn2p = bn1p + 32;                        // 128
    float* bndp = bn2p + 128;                       // 256

    float* out = (float*)d_out;

    k_layer1<<<(NROWS + 255) / 256, 256, 0, stream>>>(x, W1, b1, pw1, msk);
    k_stats<16><<<512, 256, 0, stream>>>(pw1, part);
    k_finalize<<<1, 16, 0, stream>>>(part, 512, 16, g1, be1, bn1p);
    k_agg<16><<<(KVOX * 16 + 255) / 256, 256, 0, stream>>>(pw1, bn1p, agg1);

    k_gemm2<<<1024, 256, 0, stream>>>(pw1, agg1, msk, bn1p, W2, b2, pw2);
    k_stats<64><<<512, 256, 0, stream>>>(pw2, part);
    k_finalize<<<1, 64, 0, stream>>>(part, 512, 64, g2, be2, bn2p);
    k_agg<64><<<(KVOX * 64 + 255) / 256, 256, 0, stream>>>(pw2, bn2p, agg2);

    k_gemm3_stats<<<1024, 256, 0, stream>>>(pw2, agg2, msk, bn2p, Wd, bd, part);
    k_finalize<<<1, 128, 0, stream>>>(part, 1024, 128, gd, bed, bndp);
    k_gemm3_out<<<1024, 256, 0, stream>>>(pw2, agg2, msk, bn2p, Wd, bd, bndp, out);

    k_coord<<<(KVOX * 4 + 255) / 256, 256, 0, stream>>>(crd, (int*)d_out + (size_t)128 * KVOX);
}
